// QuantumFeedForward_65481071405665
// MI455X (gfx1250) — compile-verified, matrix operand order checked by reading, not run on hardware
//
#include <hip/hip_runtime.h>

// Problem sizes (fixed by the reference)
#define EMBED   512          // C
#define FFN     2048         // F
#define BM      64           // rows (B*T) per workgroup
#define BF      32           // F-chunk per iteration
#define NCH     (FFN / BF)   // 64 chunks
#define QPAD    520          // 512 + 8 halves padding (bank-conflict-free b128)
#define WPAD    40           // 32 + 8 halves padding

typedef __attribute__((ext_vector_type(16))) __bf16        v16bf;
typedef __attribute__((ext_vector_type(8)))  float         v8f;
typedef __attribute__((ext_vector_type(4)))  unsigned int  u32x4;
typedef __attribute__((ext_vector_type(8)))  unsigned int  u32x8;

__device__ __forceinline__ unsigned short f2bf(float f) {
    union { float f; unsigned int u; } c; c.f = f;
    unsigned int u = c.u;
    return (unsigned short)((u + 0x7FFFu + ((u >> 16) & 1u)) >> 16); // RNE
}

// ---- Tensor Data Mover: 2D tile, global -> LDS, tracked by TENSORcnt ----
// D# group0: [1:0]=count=1, [63:32]=lds_addr, [120:64]=global_addr, [127:126]=type=2
// D# group1: data_size=2B, pad_enable, pad_interval/amount, dims/strides in elems.
// Groups 2/3 omitted (NULL) -> 2D tensor. LDS byte offset = low 32 bits of the
// flat address (ISA 10.2).
__device__ __forceinline__ void tdm_load_2d(unsigned lds_addr, const void* gaddr,
                                            unsigned g1w0,
                                            unsigned tdim0, unsigned tdim1,
                                            unsigned tiled0, unsigned tiled1,
                                            unsigned stride0) {
    unsigned long long ga = (unsigned long long)gaddr;
    u32x4 g0 = { 1u,                                   // count=1, user mode
                 lds_addr,                             // lds_addr
                 (unsigned)ga,                         // global_addr[31:0]
                 ((unsigned)(ga >> 32) & 0x01FFFFFFu) | 0x80000000u }; // ga[56:32] | type=2
    u32x8 g1 = { g1w0,                                 // mask=0|data_size=1|pad cfg
                 (tdim0 & 0xFFFFu) << 16,              // abar=0 | tensor_dim0[15:0]
                 (tdim0 >> 16) | ((tdim1 & 0xFFFFu) << 16), // tensor_dim0_hi | dim1_lo
                 (tdim1 >> 16) | (tiled0 << 16),       // tensor_dim1_hi | tile_dim0
                 tiled1,                               // tile_dim1 | tile_dim2=0
                 stride0,                              // tensor_dim0_stride[31:0]
                 0u, 0u };                             // stride0_hi | stride1 (unused 2D)
    asm volatile("tensor_load_to_lds %0, %1" :: "s"(g0), "s"(g1) : "memory");
}
// g1 word0 for W1 chunk: data_size=1(2B)<<16 | pad_en<<20 | interval=7(256DW)<<22 | amount=3(4DW)<<25
#define G1W0_W1 0x07D10000u
// g1 word0 for W2 chunk: data_size=1(2B)<<16 | pad_en<<20 | interval=3(16DW)<<22 | amount=3(4DW)<<25
#define G1W0_W2 0x06D10000u

// Load one 16x32 bf16 WMMA fragment (A-layout; identical pattern serves B when
// memory holds W^T rows, i.e. row index = N, K contiguous within the row).
// Per ISA 7.12.2: lane (l&15) = row, halves [k0 + (l>>4)*8 .. +7] in v[0..7],
// halves [k0+16+(l>>4)*8 .. +7] in v[8..15]  -> two ds_load_b128.
__device__ __forceinline__ v16bf load_frag16(const unsigned short* rowp, int k0, int lhi) {
    union { uint4 u[2]; v16bf v; } c;
    c.u[0] = *(const uint4*)(rowp + k0 + lhi * 8);
    c.u[1] = *(const uint4*)(rowp + k0 + 16 + lhi * 8);
    return c.v;
}

// ---- Prologue 1: cos(sum(theta)) -> wsF[0] -------------------------------
__global__ __launch_bounds__(256) void prep_theta(const float* __restrict__ theta,
                                                  float* __restrict__ wsF) {
    __shared__ float red[256];
    int t = threadIdx.x;
    red[t] = theta[t] + theta[t + 256];
    __syncthreads();
    #pragma unroll
    for (int s = 128; s > 0; s >>= 1) {
        if (t < s) red[t] += red[t + s];
        __syncthreads();
    }
    if (t == 0) wsF[0] = __cosf(red[0]);
}

// ---- Prologue 2: convert W1, W2 f32 -> bf16 in workspace -----------------
__global__ __launch_bounds__(256) void prep_weights(const float* __restrict__ W1,
                                                    const float* __restrict__ W2,
                                                    unsigned short* __restrict__ W1bf,
                                                    unsigned short* __restrict__ W2bf) {
    int i = blockIdx.x * 256 + threadIdx.x;   // 4096 blocks * 256 = 1048576 = F*C
    W1bf[i] = f2bf(W1[i]);
    W2bf[i] = f2bf(W2[i]);
}

// ---- Main fused FFN kernel ----------------------------------------------
// grid 512 x 256 threads (8 waves). Each WG: rows [blk*64, blk*64+64), all 512 cols.
__global__ __launch_bounds__(256) void ffn_fused(const float* __restrict__ x,
                                                 const float* __restrict__ wsF,
                                                 const unsigned short* __restrict__ W1bf,
                                                 const unsigned short* __restrict__ W2bf,
                                                 float* __restrict__ out) {
    __shared__ __align__(16) unsigned short qA [BM * QPAD];        //  66560 B  q tile (bf16)
    __shared__ __align__(16) unsigned short W1c[2][BF * QPAD];     // 2x33280 B  W1 chunk rows=f
    __shared__ __align__(16) unsigned short W2c[2][EMBED * WPAD];  // 2x40960 B  W2 chunk rows=c
    __shared__ __align__(16) unsigned short hA [BM * WPAD];        //   5120 B  relu(h) chunk

    const int tid  = threadIdx.x;
    const int wave = tid >> 5;
    const int lane = tid & 31;
    const int lrow = lane & 15;
    const int lhi  = lane >> 4;
    const int blk  = blockIdx.x;
    const float cosT = wsF[0];

    // TDM prefetch of one F-chunk; wave 0 issues the two descriptors.
    // W1 chunk: 32 rows x 512 elems, row stride 512  -> LDS rows padded to 520.
    // W2 chunk: 512 rows x 32 elems,  row stride 2048 -> LDS rows padded to 40.
    auto prefetch_chunk = [&](int fc, unsigned short* w1d, unsigned short* w2d) {
        if (wave == 0) {
            tdm_load_2d((unsigned)(unsigned long long)(void*)w1d,
                        W1bf + (size_t)fc * BF * EMBED,
                        G1W0_W1, EMBED, BF, EMBED, BF, EMBED);
            tdm_load_2d((unsigned)(unsigned long long)(void*)w2d,
                        W2bf + (size_t)fc * BF,
                        G1W0_W2, BF, EMBED, BF, EMBED, FFN);
        }
    };

    // Kick off chunk 0 DMA before staging q (overlaps with the cos work).
    prefetch_chunk(0, W1c[0], W2c[0]);

    // Stage q tile: q = cos(sum theta) * cos(x), converted to bf16.
    {
        const float4* xs = (const float4*)(x + (size_t)blk * BM * EMBED);
        #pragma unroll
        for (int it = 0; it < 32; ++it) {
            int i   = tid + it * 256;          // 8192 float4 total
            int row = i >> 7;                  // 128 float4 per row
            int q   = i & 127;
            float4 v = xs[row * 128 + q];
            uint2 pk;
            pk.x = (unsigned)f2bf(cosT * __cosf(v.x)) |
                   ((unsigned)f2bf(cosT * __cosf(v.y)) << 16);
            pk.y = (unsigned)f2bf(cosT * __cosf(v.z)) |
                   ((unsigned)f2bf(cosT * __cosf(v.w)) << 16);
            *(uint2*)&qA[row * QPAD + q * 4] = pk;   // 8B aligned
        }
    }

    // Persistent f32 accumulators: wave owns cols [wave*64, wave*64+64):
    // 4 M-tiles x 4 N-tiles of 16x16 = 128 VGPRs.
    v8f acc[4][4];
    #pragma unroll
    for (int a = 0; a < 4; ++a)
        #pragma unroll
        for (int b = 0; b < 4; ++b) acc[a][b] = (v8f)(0.f);

    const int tm1 = wave >> 1;   // stage-1: this wave's M tile (0..3)
    const int tf1 = wave & 1;    // stage-1: this wave's F tile (0..1)

    if (wave == 0) __builtin_amdgcn_s_wait_tensorcnt(0);
    __syncthreads();             // qA + chunk 0 visible to all waves

    for (int fc = 0; fc < NCH; ++fc) {
        const int cur = fc & 1, nxt = cur ^ 1;
        if (fc + 1 < NCH)        // overlap next chunk's TDM with this chunk's WMMAs
            prefetch_chunk(fc + 1, W1c[nxt], W2c[nxt]);

        // ---- stage 1: h = relu(q @ W1^T) for this F-chunk, K = 512 ----
        // split-K dual accumulator chains for XDL pipeline overlap
        v8f h0 = (v8f)(0.f), h1 = (v8f)(0.f);
        const unsigned short* arow = &qA [(tm1 * 16 + lrow) * QPAD];
        const unsigned short* brow = &W1c[cur][(tf1 * 16 + lrow) * QPAD];
        #pragma unroll
        for (int k0 = 0; k0 < EMBED / 2; k0 += 32) {
            v16bf a0 = load_frag16(arow, k0, lhi);
            v16bf b0 = load_frag16(brow, k0, lhi);
            h0 = __builtin_amdgcn_wmma_f32_16x16x32_bf16(false, a0, false, b0,
                                                         (short)0, h0, false, false);
            v16bf a1 = load_frag16(arow, k0 + EMBED / 2, lhi);
            v16bf b1 = load_frag16(brow, k0 + EMBED / 2, lhi);
            h1 = __builtin_amdgcn_wmma_f32_16x16x32_bf16(false, a1, false, b1,
                                                         (short)0, h1, false, false);
        }
        v8f h = h0 + h1;

        // relu -> bf16 -> hA (C/D layout: VGPR g holds row lhi*8+g, col lrow)
        #pragma unroll
        for (int g = 0; g < 8; ++g) {
            float v = h[g];
            v = v > 0.f ? v : 0.f;
            hA[(tm1 * 16 + lhi * 8 + g) * WPAD + tf1 * 16 + lrow] = f2bf(v);
        }
        __syncthreads();         // hA visible

        // ---- stage 2: out += h_chunk @ W2^T (K = 32, one WMMA step) ----
        v16bf aF[4];
        #pragma unroll
        for (int tm = 0; tm < 4; ++tm)
            aF[tm] = load_frag16(&hA[(tm * 16 + lrow) * WPAD], 0, lhi);
        #pragma unroll
        for (int tn = 0; tn < 4; ++tn) {
            v16bf bF = load_frag16(&W2c[cur][(wave * 64 + tn * 16 + lrow) * WPAD], 0, lhi);
            #pragma unroll
            for (int tm = 0; tm < 4; ++tm)
                acc[tm][tn] = __builtin_amdgcn_wmma_f32_16x16x32_bf16(
                    false, aF[tm], false, bF, (short)0, acc[tm][tn], false, false);
        }

        if (wave == 0) __builtin_amdgcn_s_wait_tensorcnt(0); // own TDM done
        __syncthreads();         // all waves see new chunk; cur buffers + hA freed
    }

    // ---- epilogue: store 64x512 f32 tile ----
    #pragma unroll
    for (int tm = 0; tm < 4; ++tm)
        #pragma unroll
        for (int tn = 0; tn < 4; ++tn) {
            int col = wave * 64 + tn * 16 + lrow;
            #pragma unroll
            for (int g = 0; g < 8; ++g) {
                int row = blk * BM + tm * 16 + lhi * 8 + g;
                out[(size_t)row * EMBED + col] = acc[tm][tn][g];
            }
        }
}

extern "C" void kernel_launch(void* const* d_in, const int* in_sizes, int n_in,
                              void* d_out, int out_size, void* d_ws, size_t ws_size,
                              hipStream_t stream) {
    const float* x     = (const float*)d_in[0];   // [8,4096,512]
    const float* theta = (const float*)d_in[1];   // [512]
    const float* W1    = (const float*)d_in[2];   // [2048,512]
    const float* W2    = (const float*)d_in[3];   // [512,2048]
    float* out = (float*)d_out;

    // workspace: [0..15] scalar, then W1 bf16 (2 MB), W2 bf16 (2 MB)
    float* wsF = (float*)d_ws;
    unsigned short* W1bf = (unsigned short*)((char*)d_ws + 16);
    unsigned short* W2bf = W1bf + (size_t)FFN * EMBED;

    prep_theta  <<<1,    256, 0, stream>>>(theta, wsF);
    prep_weights<<<4096, 256, 0, stream>>>(W1, W2, W1bf, W2bf);
    ffn_fused   <<<512,  256, 0, stream>>>(x, wsF, W1bf, W2bf, out);
}
